// bottle_view_graph_54030688584323
// MI455X (gfx1250) — compile-verified, matrix operand order checked by reading, not run on hardware
//
#include <hip/hip_runtime.h>
#include <hip/hip_bf16.h>

// ---------------------------------------------------------------------------
// CDNA5 (gfx1250, wave32) implementation of the GNN + cross-attention model.
// All dense GEMMs go through v_wmma_f32_16x16x32_f16 (f16 in, f32 accum).
// Attention uses a flash-style tile kernel (P staged via LDS -> ds ops).
// Segment softmax over edges uses ordered-uint atomicMax + float atomicAdd.
// ---------------------------------------------------------------------------

typedef __attribute__((ext_vector_type(16))) _Float16 v16h;
typedef __attribute__((ext_vector_type(8)))  float    v8f;

#define HDIM 128

// K-index map for 16-bit A/B fragments (CDNA5 ISA 7.12.2):
// lane group g = lane>>4; element i -> K = (i&8)*2 + (i&7) + 8*g
__device__ __forceinline__ int kmap(int i, int g) {
  return ((i & 8) << 1) + (i & 7) + (g << 3);
}

// ---------------------------------------------------------------------------
// Generic WMMA GEMM: C[M,N] = act(A[M,K] @ W[K,N] + bias)
// grid = (M/16, N/16), block = 32 (one wave per 16x16 tile).
// ACT: 0 = none, 1 = SiLU, 2 = LeakyReLU(0.01)
// ---------------------------------------------------------------------------
template <int ACT>
__global__ void gemm_kernel(const float* __restrict__ A, int lda,
                            const float* __restrict__ W, int ldb,
                            const float* __restrict__ bias,
                            float* __restrict__ C, int ldc, int K) {
  const int lane = threadIdx.x & 31;
  const int g    = lane >> 4;
  const int ln   = lane & 15;
  const long m   = (long)blockIdx.x * 16 + ln;   // A row for fragment
  const long n   = (long)blockIdx.y * 16 + ln;   // B / C column
  v8f acc = {0.f, 0.f, 0.f, 0.f, 0.f, 0.f, 0.f, 0.f};
  for (int k0 = 0; k0 < K; k0 += 32) {
    v16h a, b;
#pragma unroll
    for (int i = 0; i < 16; ++i) {
      int k = k0 + kmap(i, g);
      a[i] = (k < K) ? (_Float16)A[m * lda + k]          : (_Float16)0.f;
      b[i] = (k < K) ? (_Float16)W[(long)k * ldb + n]    : (_Float16)0.f;
    }
    acc = __builtin_amdgcn_wmma_f32_16x16x32_f16(false, a, false, b,
                                                 (short)0, acc, false, false);
  }
#pragma unroll
  for (int r = 0; r < 8; ++r) {
    long mm = (long)blockIdx.x * 16 + r + (g << 3);
    float v = acc[r];
    if (bias) v += bias[n];
    if (ACT == 1)      v = v / (1.f + __expf(-v));     // SiLU
    else if (ACT == 2) v = (v > 0.f) ? v : 0.01f * v;  // LeakyReLU 0.01
    C[mm * ldc + n] = v;
  }
}

// ---------------------------------------------------------------------------
// Flash-attention tile kernel: one wave per (b*NH, 16-query tile).
// Q,K,V,O are [B*L, 128] with the head slice at column head*32.
// Online softmax; P staged through LDS (f16) to re-layout C->A fragment.
// ---------------------------------------------------------------------------
__global__ void flash_attn_kernel(const float* __restrict__ Q,
                                  const float* __restrict__ Km,
                                  const float* __restrict__ Vm,
                                  float* __restrict__ O, int Lq, int Lk) {
  __shared__ _Float16 sP[16 * 32];
  const int lane = threadIdx.x & 31;
  const int g    = lane >> 4;
  const int ln   = lane & 15;
  const int bh   = blockIdx.x;
  const int b    = bh >> 2;     // NH = 4
  const int head = bh & 3;
  const int hc   = head * 32;
  const float scale = 0.17677669529663687f;  // 1/sqrt(32)

  const long qbase = (long)b * Lq + (long)blockIdx.y * 16;
  const long kbase = (long)b * Lk;

  v16h aQ;
#pragma unroll
  for (int i = 0; i < 16; ++i) {
    int d = kmap(i, g);
    aQ[i] = (_Float16)Q[(qbase + ln) * HDIM + hc + d];
  }

  v8f O0 = {0.f, 0.f, 0.f, 0.f, 0.f, 0.f, 0.f, 0.f};
  v8f O1 = {0.f, 0.f, 0.f, 0.f, 0.f, 0.f, 0.f, 0.f};
  float rmax[8], rsum[8];
#pragma unroll
  for (int r = 0; r < 8; ++r) { rmax[r] = -1e30f; rsum[r] = 0.f; }

  for (int kt = 0; kt < Lk; kt += 32) {
    v16h bK0, bK1;
#pragma unroll
    for (int i = 0; i < 16; ++i) {
      int d = kmap(i, g);
      bK0[i] = (_Float16)Km[(kbase + kt + ln)      * HDIM + hc + d];
      bK1[i] = (_Float16)Km[(kbase + kt + 16 + ln) * HDIM + hc + d];
    }
    v8f Z = {0.f, 0.f, 0.f, 0.f, 0.f, 0.f, 0.f, 0.f};
    v8f S0 = __builtin_amdgcn_wmma_f32_16x16x32_f16(false, aQ, false, bK0,
                                                    (short)0, Z, false, false);
    v8f S1 = __builtin_amdgcn_wmma_f32_16x16x32_f16(false, aQ, false, bK1,
                                                    (short)0, Z, false, false);
#pragma unroll
    for (int r = 0; r < 8; ++r) {
      float s0 = S0[r] * scale, s1 = S1[r] * scale;
      float t = fmaxf(s0, s1);
#pragma unroll
      for (int off = 8; off > 0; off >>= 1) t = fmaxf(t, __shfl_xor(t, off, 32));
      float mnew = fmaxf(rmax[r], t);
      float p0 = __expf(s0 - mnew), p1 = __expf(s1 - mnew);
      float f  = __expf(rmax[r] - mnew);
      float ps = p0 + p1;
#pragma unroll
      for (int off = 8; off > 0; off >>= 1) ps += __shfl_xor(ps, off, 32);
      rsum[r] = rsum[r] * f + ps;
      rmax[r] = mnew;
      O0[r] *= f; O1[r] *= f;
      int m = r + (g << 3);
      sP[m * 32 + ln]      = (_Float16)p0;
      sP[m * 32 + 16 + ln] = (_Float16)p1;
    }
    v16h aP, bV0, bV1;
#pragma unroll
    for (int i = 0; i < 16; ++i) {
      int kk = kmap(i, g);
      aP[i]  = sP[ln * 32 + kk];
      bV0[i] = (_Float16)Vm[(kbase + kt + kk) * HDIM + hc + ln];
      bV1[i] = (_Float16)Vm[(kbase + kt + kk) * HDIM + hc + 16 + ln];
    }
    O0 = __builtin_amdgcn_wmma_f32_16x16x32_f16(false, aP, false, bV0,
                                                (short)0, O0, false, false);
    O1 = __builtin_amdgcn_wmma_f32_16x16x32_f16(false, aP, false, bV1,
                                                (short)0, O1, false, false);
  }
#pragma unroll
  for (int r = 0; r < 8; ++r) {
    int m = r + (g << 3);
    float inv = 1.f / rsum[r];
    O[(qbase + m) * HDIM + hc + ln]      = O0[r] * inv;
    O[(qbase + m) * HDIM + hc + 16 + ln] = O1[r] * inv;
  }
}

// ---------------------------------------------------------------------------
// GATv2 edge pipeline kernels
// ---------------------------------------------------------------------------
__device__ __forceinline__ unsigned enc_f(float f) {
  unsigned b = __float_as_uint(f);
  return b ^ ((b >> 31) ? 0xFFFFFFFFu : 0x80000000u);
}
__device__ __forceinline__ float dec_f(unsigned u) {
  unsigned b = u ^ ((u >> 31) ? 0x80000000u : 0xFFFFFFFFu);
  return __uint_as_float(b);
}

__global__ void fill_f32_kernel(float* p, float v, int n) {
  int i = blockIdx.x * 256 + threadIdx.x;
  if (i < n) p[i] = v;
}
__global__ void fill_u32_kernel(unsigned* p, unsigned v, int n) {
  int i = blockIdx.x * 256 + threadIdx.x;
  if (i < n) p[i] = v;
}
__global__ void fill_bias_kernel(float* p, const float* __restrict__ b, long n) {
  long i = (long)blockIdx.x * 256 + threadIdx.x;
  if (i < n) p[i] = b[i & (HDIM - 1)];
}

// one wave per edge: logit[e] = att . leaky_relu(xl[src]+xr[dst]+we[e], 0.2)
__global__ void edge_logit_kernel(const float* __restrict__ xl,
                                  const float* __restrict__ xr,
                                  const float* __restrict__ we,
                                  const int* __restrict__ src,
                                  const int* __restrict__ dst,
                                  const float* __restrict__ att,
                                  float* __restrict__ logit, int E) {
  int e = blockIdx.x * 8 + (threadIdx.x >> 5);
  int lane = threadIdx.x & 31;
  if (e >= E) return;
  long s = src[e], d = dst[e];
  float sum = 0.f;
#pragma unroll
  for (int j = 0; j < 4; ++j) {
    int h = lane + 32 * j;
    float m = xl[s * HDIM + h] + xr[d * HDIM + h] + we[(long)e * HDIM + h];
    m = (m > 0.f) ? m : 0.2f * m;
    sum += m * att[h];
  }
#pragma unroll
  for (int off = 16; off > 0; off >>= 1) sum += __shfl_xor(sum, off, 32);
  if (lane == 0) logit[e] = sum;
}

__global__ void seg_max_kernel(const float* logit, const int* dst,
                               unsigned* segmax, int E) {
  int e = blockIdx.x * 256 + threadIdx.x;
  if (e < E) atomicMax(&segmax[dst[e]], enc_f(logit[e]));
}

__global__ void seg_expsum_kernel(const float* logit, const int* dst,
                                  const unsigned* segmax, float* w, float* den,
                                  int E) {
  int e = blockIdx.x * 256 + threadIdx.x;
  if (e < E) {
    float mx = dec_f(segmax[dst[e]]);
    float ww = __expf(logit[e] - mx);
    w[e] = ww;
    atomicAdd(&den[dst[e]], ww);
  }
}

// one wave per edge: out[dst] += alpha * xl[src]  (out pre-filled with bias)
__global__ void scatter_kernel(const float* __restrict__ xl,
                               const int* __restrict__ src,
                               const int* __restrict__ dst,
                               const float* __restrict__ w,
                               const float* __restrict__ den,
                               float* __restrict__ out, int E) {
  int e = blockIdx.x * 8 + (threadIdx.x >> 5);
  int lane = threadIdx.x & 31;
  if (e >= E) return;
  long s = src[e], d = dst[e];
  float alpha = w[e] / (den[d] + 1e-16f);
#pragma unroll
  for (int j = 0; j < 4; ++j) {
    int h = lane + 32 * j;
    atomicAdd(&out[d * HDIM + h], alpha * xl[s * HDIM + h]);
  }
}

// ---------------------------------------------------------------------------
// LayerNorm(x + resid) * g + b  — one wave per row of 128
// ---------------------------------------------------------------------------
__global__ void ln_residual_kernel(const float* __restrict__ x,
                                   const float* __restrict__ resid,
                                   const float* __restrict__ gam,
                                   const float* __restrict__ bet,
                                   float* __restrict__ out, int N) {
  int row = blockIdx.x * 4 + (threadIdx.x >> 5);
  int lane = threadIdx.x & 31;
  if (row >= N) return;
  long base = (long)row * HDIM;
  float v[4];
  float mu = 0.f;
#pragma unroll
  for (int j = 0; j < 4; ++j) {
    int h = lane + 32 * j;
    v[j] = x[base + h] + resid[base + h];
    mu += v[j];
  }
#pragma unroll
  for (int off = 16; off > 0; off >>= 1) mu += __shfl_xor(mu, off, 32);
  mu *= (1.f / HDIM);
  float var = 0.f;
#pragma unroll
  for (int j = 0; j < 4; ++j) { float d = v[j] - mu; var += d * d; }
#pragma unroll
  for (int off = 16; off > 0; off >>= 1) var += __shfl_xor(var, off, 32);
  var *= (1.f / HDIM);
  float rstd = rsqrtf(var + 1e-5f);
#pragma unroll
  for (int j = 0; j < 4; ++j) {
    int h = lane + 32 * j;
    out[base + h] = (v[j] - mu) * rstd * gam[h] + bet[h];
  }
}

// mean pool over Lrows rows per batch into pooled[b*256 + col_off + c]
__global__ void pool_kernel(const float* __restrict__ d, int Lrows,
                            float* __restrict__ pooled, int col_off) {
  int b = blockIdx.x;
  int c = threadIdx.x;  // 128 threads
  float s = 0.f;
  long base = (long)b * Lrows * HDIM + c;
  for (int r = 0; r < Lrows; ++r) s += d[base + (long)r * HDIM];
  pooled[b * 256 + col_off + c] = s / (float)Lrows;
}

// out[b] = dot(fch[b,:256], W2[:,0]) + b2
__global__ void fc2_kernel(const float* __restrict__ fch,
                           const float* __restrict__ W2,
                           const float* __restrict__ b2,
                           float* __restrict__ out) {
  int b = blockIdx.x;
  int lane = threadIdx.x & 31;
  float s = 0.f;
#pragma unroll
  for (int j = 0; j < 8; ++j) {
    int c = lane + 32 * j;
    s += fch[b * 256 + c] * W2[c];
  }
#pragma unroll
  for (int off = 16; off > 0; off >>= 1) s += __shfl_xor(s, off, 32);
  if (lane == 0) out[b] = s + b2[0];
}

// ---------------------------------------------------------------------------
// Host side
// ---------------------------------------------------------------------------
static void gat_layer(hipStream_t s, const float* hin, float* hout, float* xl,
                      float* xr, float* we, const float* eattr, int ED,
                      const int* src, const int* dst, int E, int N,
                      const float* Wl, const float* Wr, const float* We,
                      const float* att, const float* bias, float* logit,
                      float* w, unsigned* segmax, float* den) {
  dim3 blk(32);
  gemm_kernel<0><<<dim3(N / 16, HDIM / 16), blk, 0, s>>>(hin, HDIM, Wl, HDIM,
                                                         nullptr, xl, HDIM, HDIM);
  gemm_kernel<0><<<dim3(N / 16, HDIM / 16), blk, 0, s>>>(hin, HDIM, Wr, HDIM,
                                                         nullptr, xr, HDIM, HDIM);
  gemm_kernel<0><<<dim3(E / 16, HDIM / 16), blk, 0, s>>>(eattr, ED, We, HDIM,
                                                         nullptr, we, HDIM, ED);
  fill_u32_kernel<<<(N + 255) / 256, 256, 0, s>>>(segmax, 0u, N);
  fill_f32_kernel<<<(N + 255) / 256, 256, 0, s>>>(den, 0.f, N);
  fill_bias_kernel<<<((long)N * HDIM + 255) / 256, 256, 0, s>>>(hout, bias,
                                                                (long)N * HDIM);
  edge_logit_kernel<<<(E + 7) / 8, 256, 0, s>>>(xl, xr, we, src, dst, att,
                                                logit, E);
  seg_max_kernel<<<(E + 255) / 256, 256, 0, s>>>(logit, dst, segmax, E);
  seg_expsum_kernel<<<(E + 255) / 256, 256, 0, s>>>(logit, dst, segmax, w, den, E);
  scatter_kernel<<<(E + 7) / 8, 256, 0, s>>>(xl, src, dst, w, den, hout, E);
}

extern "C" void kernel_launch(void* const* d_in, const int* in_sizes, int n_in,
                              void* d_out, int out_size, void* d_ws,
                              size_t ws_size, hipStream_t stream) {
  (void)in_sizes; (void)n_in; (void)out_size; (void)ws_size;
  const int NP = 65536, NL = 8192, EP = 262144, EL = 32768;
  const size_t NPH = (size_t)NP * HDIM, NLH = (size_t)NL * HDIM;
  const size_t EPH = (size_t)EP * HDIM, ELH = (size_t)EL * HDIM;

  const float* x_lig = (const float*)d_in[0];
  const float* x_pro = (const float*)d_in[1];
  const float* lea   = (const float*)d_in[2];
  const float* pea   = (const float*)d_in[3];
  const float* winlW = (const float*)d_in[4];
  const float* winlB = (const float*)d_in[5];
  const float* winpW = (const float*)d_in[6];
  const float* winpB = (const float*)d_in[7];
  const int GL[2] = {8, 13};   // gat_lig layers: Wl,Wr,We,att,b
  const int GP[2] = {18, 23};  // gat_pro layers
  const float* qW = (const float*)d_in[28]; const float* qB = (const float*)d_in[29];
  const float* kW = (const float*)d_in[30]; const float* kB = (const float*)d_in[31];
  const float* vW = (const float*)d_in[32]; const float* vB = (const float*)d_in[33];
  const float* oW = (const float*)d_in[34]; const float* oB = (const float*)d_in[35];
  const float* mrW0 = (const float*)d_in[36]; const float* mrB0 = (const float*)d_in[37];
  const float* mrW1 = (const float*)d_in[38]; const float* mrB1 = (const float*)d_in[39];
  const float* mlW0 = (const float*)d_in[40]; const float* mlB0 = (const float*)d_in[41];
  const float* mlW1 = (const float*)d_in[42]; const float* mlB1 = (const float*)d_in[43];
  const float* lnaG = (const float*)d_in[44]; const float* lnaB = (const float*)d_in[45];
  const float* lnlG = (const float*)d_in[46]; const float* lnlB = (const float*)d_in[47];
  const float* fcW0 = (const float*)d_in[48]; const float* fcB0 = (const float*)d_in[49];
  const float* fcW1 = (const float*)d_in[50]; const float* fcB1 = (const float*)d_in[51];
  const int* lsrc = (const int*)d_in[52]; const int* ldst = (const int*)d_in[53];
  const int* psrc = (const int*)d_in[54]; const int* pdst = (const int*)d_in[55];
  float* out = (float*)d_out;

  // Workspace carve-up (~306 MB)
  float* ws = (float*)d_ws;
  float* hpro_a = ws;
  float* hpro_b = hpro_a + NPH;
  float* xlp    = hpro_b + NPH;
  float* xrp    = xlp + NPH;
  float* bigp   = xrp + NPH;        // EPH floats = 4*NPH: we_pro / q,k,v,ao
  float* hlig_a = bigp + EPH;
  float* hlig_b = hlig_a + NLH;
  float* xll    = hlig_b + NLH;
  float* xrl    = xll + NLH;
  float* bigl   = xrl + NLH;        // ELH floats = 4*NLH
  float* logitp = bigl + ELH;
  float* wp     = logitp + EP;
  unsigned* segmaxp = (unsigned*)(wp + EP);
  float* denp   = (float*)(segmaxp + NP);
  float* logitl = denp + NP;
  float* wl     = logitl + EL;
  unsigned* segmaxl = (unsigned*)(wl + EL);
  float* denl   = (float*)(segmaxl + NL);
  float* pooled = denl + NL;        // 64*256
  float* fch    = pooled + 64 * 256;

  dim3 blk(32);

  // ---- input projections (SiLU) ----
  gemm_kernel<1><<<dim3(NP / 16, 8), blk, 0, stream>>>(x_pro, 64, winpW, HDIM,
                                                       winpB, hpro_a, HDIM, 64);
  gemm_kernel<1><<<dim3(NL / 16, 8), blk, 0, stream>>>(x_lig, 64, winlW, HDIM,
                                                       winlB, hlig_a, HDIM, 64);

  // ---- GATv2 layers (2 each, ping-pong a<->b, final lands in *_a) ----
  {
    float* in = hlig_a; float* outb = hlig_b;
    for (int l = 0; l < 2; ++l) {
      gat_layer(stream, in, outb, xll, xrl, bigl, lea, 16, lsrc, ldst, EL, NL,
                (const float*)d_in[GL[l] + 0], (const float*)d_in[GL[l] + 1],
                (const float*)d_in[GL[l] + 2], (const float*)d_in[GL[l] + 3],
                (const float*)d_in[GL[l] + 4], logitl, wl, segmaxl, denl);
      float* t = in; in = outb; outb = t;
    }
  }
  {
    float* in = hpro_a; float* outb = hpro_b;
    for (int l = 0; l < 2; ++l) {
      gat_layer(stream, in, outb, xlp, xrp, bigp, pea, 32, psrc, pdst, EP, NP,
                (const float*)d_in[GP[l] + 0], (const float*)d_in[GP[l] + 1],
                (const float*)d_in[GP[l] + 2], (const float*)d_in[GP[l] + 3],
                (const float*)d_in[GP[l] + 4], logitp, wp, segmaxp, denp);
      float* t = in; in = outb; outb = t;
    }
  }

  // ---- cross-attention (shared QKV/O weights, both directions) ----
  float* qpro = bigp;           float* kpro = bigp + NPH;
  float* vpro = bigp + 2 * NPH; float* aopro = bigp + 3 * NPH;
  float* qlig = bigl;           float* klig = bigl + NLH;
  float* vlig = bigl + 2 * NLH; float* aolig = bigl + 3 * NLH;

  gemm_kernel<0><<<dim3(NP / 16, 8), blk, 0, stream>>>(hpro_a, HDIM, qW, HDIM, qB, qpro, HDIM, HDIM);
  gemm_kernel<0><<<dim3(NP / 16, 8), blk, 0, stream>>>(hpro_a, HDIM, kW, HDIM, kB, kpro, HDIM, HDIM);
  gemm_kernel<0><<<dim3(NP / 16, 8), blk, 0, stream>>>(hpro_a, HDIM, vW, HDIM, vB, vpro, HDIM, HDIM);
  gemm_kernel<0><<<dim3(NL / 16, 8), blk, 0, stream>>>(hlig_a, HDIM, qW, HDIM, qB, qlig, HDIM, HDIM);
  gemm_kernel<0><<<dim3(NL / 16, 8), blk, 0, stream>>>(hlig_a, HDIM, kW, HDIM, kB, klig, HDIM, HDIM);
  gemm_kernel<0><<<dim3(NL / 16, 8), blk, 0, stream>>>(hlig_a, HDIM, vW, HDIM, vB, vlig, HDIM, HDIM);

  // aa_att: queries = pro (Lq=1024), keys/vals = lig (Lk=128)
  flash_attn_kernel<<<dim3(64 * 4, 1024 / 16), blk, 0, stream>>>(qpro, klig, vlig, aopro, 1024, 128);
  // lig_att: queries = lig (Lq=128), keys/vals = pro (Lk=1024)
  flash_attn_kernel<<<dim3(64 * 4, 128 / 16), blk, 0, stream>>>(qlig, kpro, vpro, aolig, 128, 1024);

  // output projections -> xlp / xll
  gemm_kernel<0><<<dim3(NP / 16, 8), blk, 0, stream>>>(aopro, HDIM, oW, HDIM, oB, xlp, HDIM, HDIM);
  gemm_kernel<0><<<dim3(NL / 16, 8), blk, 0, stream>>>(aolig, HDIM, oW, HDIM, oB, xll, HDIM, HDIM);

  // ---- MLP + residual + LayerNorm ----
  gemm_kernel<2><<<dim3(NP / 16, 8), blk, 0, stream>>>(xlp, HDIM, mrW0, HDIM, mrB0, xrp, HDIM, HDIM);
  gemm_kernel<2><<<dim3(NP / 16, 8), blk, 0, stream>>>(xrp, HDIM, mrW1, HDIM, mrB1, qpro, HDIM, HDIM);
  ln_residual_kernel<<<NP / 4, 128, 0, stream>>>(qpro, hpro_a, lnaG, lnaB, hpro_b, NP);

  gemm_kernel<2><<<dim3(NL / 16, 8), blk, 0, stream>>>(xll, HDIM, mlW0, HDIM, mlB0, xrl, HDIM, HDIM);
  gemm_kernel<2><<<dim3(NL / 16, 8), blk, 0, stream>>>(xrl, HDIM, mlW1, HDIM, mlB1, qlig, HDIM, HDIM);
  ln_residual_kernel<<<NL / 4, 128, 0, stream>>>(qlig, hlig_a, lnlG, lnlB, hlig_b, NL);

  // ---- mean pool + FC head ----
  pool_kernel<<<64, 128, 0, stream>>>(hlig_b, 128, pooled, 0);
  pool_kernel<<<64, 128, 0, stream>>>(hpro_b, 1024, pooled, 128);
  gemm_kernel<2><<<dim3(64 / 16, 256 / 16), blk, 0, stream>>>(pooled, 256, fcW0,
                                                              256, fcB0, fch, 256, 256);
  fc2_kernel<<<64, 32, 0, stream>>>(fch, fcW1, fcB1, out);
}
